// GNNModule_24386824307418
// MI455X (gfx1250) — matrix operand order repaired and checked by reading, not compiled, not run on hardware
//
#include <hip/hip_runtime.h>
#include <math.h>

typedef __attribute__((ext_vector_type(16))) _Float16 v16h;
typedef __attribute__((ext_vector_type(8)))  float    v8f;
typedef __attribute__((ext_vector_type(4)))  unsigned int u32x4;
typedef __attribute__((ext_vector_type(8)))  int      i32x8;
typedef __attribute__((ext_vector_type(4)))  int      i32x4;

#define D 128

// ---------- order-preserving float <-> uint for atomic segment-max ----------
static __device__ __forceinline__ unsigned ordf(float f) {
    unsigned u = __float_as_uint(f);
    return (u & 0x80000000u) ? ~u : (u | 0x80000000u);
}
static __device__ __forceinline__ float unordf(unsigned o) {
    return (o & 0x80000000u) ? __uint_as_float(o & 0x7fffffffu)
                             : __uint_as_float(~o);
}

// ---------- mean(edge_attr) reduction ----------
__global__ void k_ea_sum(const float* __restrict__ ea, int E, float* __restrict__ out) {
    float s = 0.0f;
    for (int i = blockIdx.x * blockDim.x + threadIdx.x; i < E; i += gridDim.x * blockDim.x)
        s += ea[i];
    for (int o = 16; o > 0; o >>= 1) s += __shfl_xor(s, o, 32);
    if ((threadIdx.x & 31) == 0) atomicAdd(out, s);
}

// ---------- f32 -> f16 activation convert ----------
__global__ void k_h2half(const float* __restrict__ h, _Float16* __restrict__ hh, int n) {
    int i = blockIdx.x * blockDim.x + threadIdx.x;
    if (i < n) hh[i] = (_Float16)h[i];
}

// ---------- weight convert + transpose: Wt[o][k] = W[k][o], f16 ----------
__global__ void k_w2half(const float* __restrict__ Wl, const float* __restrict__ Wr,
                         _Float16* __restrict__ WtL, _Float16* __restrict__ WtR) {
    int i = blockIdx.x * blockDim.x + threadIdx.x; // 0 .. D*D-1
    if (i >= D * D) return;
    int k = i >> 7, o = i & 127;
    WtL[o * D + k] = (_Float16)Wl[i];
    WtR[o * D + k] = (_Float16)Wr[i];
}

// ---------- WMMA GEMM: Y[N][128] = hh(f16) @ W + bias  (blockIdx.y: 0=L, 1=R) ----------
__global__ __launch_bounds__(256) void k_gemm(
    const _Float16* __restrict__ hh,
    const _Float16* __restrict__ WtL, const _Float16* __restrict__ WtR,
    const float* __restrict__ bl, const float* __restrict__ br,
    float* __restrict__ xl, float* __restrict__ xr) {
    const _Float16* Wt  = blockIdx.y ? WtR : WtL;
    const float*    bia = blockIdx.y ? br  : bl;
    float*          Y   = blockIdx.y ? xr  : xl;

    __shared__ __align__(16) _Float16 Wlds[D * D]; // 32 KB: full transposed weight tile

#if __has_builtin(__builtin_amdgcn_tensor_load_to_lds) && __has_builtin(__builtin_amdgcn_s_wait_tensorcnt)
    // Stage weight tile via Tensor Data Mover: one 2D descriptor, issued by wave 0.
    if (threadIdx.x < 32) {
        unsigned long long ga = (unsigned long long)(size_t)(const void*)Wt;
        u32x4 g0;
        g0[0] = 1u;                                          // count=1, user D#
        g0[1] = (unsigned)(size_t)(void*)Wlds;               // LDS byte address (flat addr[31:0])
        g0[2] = (unsigned)(ga & 0xffffffffu);                // global_addr[31:0]
        g0[3] = (unsigned)((ga >> 32) & 0x1ffffffu) | (2u << 30); // global_addr[56:32] | type=2
        i32x8 g1;
        g1[0] = (int)(1u << 16);                             // workgroup_mask=0, data_size=2B
        g1[1] = (int)((unsigned)(D & 0xffff) << 16);         // tensor_dim0[15:0]
        g1[2] = (int)(((unsigned)(D >> 16) & 0xffffu) |      // tensor_dim0[31:16]
                      ((unsigned)(D & 0xffff) << 16));       // tensor_dim1[15:0]
        g1[3] = (int)(((unsigned)(D >> 16) & 0xffffu) |      // tensor_dim1[31:16]
                      ((unsigned)D << 16));                  // tile_dim0 = 128
        g1[4] = D;                                           // tile_dim1 = 128, tile_dim2 = 0
        g1[5] = D;                                           // tensor_dim0_stride[31:0] = 128
        g1[6] = 0;                                           // stride hi / dim1_stride lo
        g1[7] = 0;
        i32x4 g2 = {};                                       // unused (<=2D tensor)
        i32x4 g3 = {};
        i32x8 g4 = {};                                       // extra group (6-arg toolchain form)
        __builtin_amdgcn_tensor_load_to_lds(g0, g1, g2, g3, g4, 0);
        __builtin_amdgcn_s_wait_tensorcnt(0);
    }
#else
    {
        const uint4* src = (const uint4*)Wt;
        uint4*       dst = (uint4*)Wlds;
        for (int i = threadIdx.x; i < D * D / 8; i += 256) dst[i] = src[i];
    }
#endif
    __syncthreads();

    const int wave    = threadIdx.x >> 5;
    const int lane    = threadIdx.x & 31;
    const int rowbase = blockIdx.x * 128 + (wave >> 1) * 32;
    const int colbase = (wave & 1) * 64;
    const int m       = lane & 15;          // A row / B column within 16
    const int kb      = (lane >> 4) * 8;    // A-fragment K sub-base (runs of 8)
    const int kb16    = (lane >> 4) * 16;   // unused (B fragment also uses kb)

    v8f zero = {};
    v8f acc[2][4];
    for (int mi = 0; mi < 2; ++mi)
        for (int ni = 0; ni < 4; ++ni) acc[mi][ni] = zero;

    union Frag { v16h v; uint4 u[2]; };

    for (int k0 = 0; k0 < D; k0 += 32) {
        Frag a[2];
        for (int mi = 0; mi < 2; ++mi) {
            const _Float16* p = hh + (size_t)(rowbase + mi * 16 + m) * D + k0 + kb;
            a[mi].u[0] = *(const uint4*)p;          // K = kb .. kb+7
            a[mi].u[1] = *(const uint4*)(p + 16);   // K = kb+16 .. kb+23
        }
        Frag b[4];
        for (int ni = 0; ni < 4; ++ni) {
            const _Float16* p = Wlds + (colbase + ni * 16 + m) * D + k0 + kb;
            b[ni].u[0] = *(const uint4*)p;          // K = kb .. kb+7
            b[ni].u[1] = *(const uint4*)(p + 16);   // K = kb+16 .. kb+23
        }
        for (int mi = 0; mi < 2; ++mi)
            for (int ni = 0; ni < 4; ++ni)
                acc[mi][ni] = __builtin_amdgcn_wmma_f32_16x16x32_f16(
                    false, a[mi].v, false, b[ni].v, (short)0, acc[mi][ni], false, false);
    }

    const int rsub = (lane >> 4) * 8;
    for (int mi = 0; mi < 2; ++mi)
        for (int ni = 0; ni < 4; ++ni) {
            int   col = colbase + ni * 16 + m;
            float bv  = bia[col];
            for (int r = 0; r < 8; ++r) {
                int row = rowbase + mi * 16 + rsub + r;
                Y[(size_t)row * D + col] = acc[mi][ni][r] + bv;
            }
        }
}

// ---------- edge pass A: attention logits + segment max (wave per edge) ----------
__global__ __launch_bounds__(256) void k_edge_a(
    const float* __restrict__ xl, const float* __restrict__ xr,
    const int* __restrict__ esrc, const int* __restrict__ edst,
    const float* __restrict__ eattr, const float* __restrict__ easum,
    const float* __restrict__ We, const float* __restrict__ att,
    float* __restrict__ aBuf, unsigned* __restrict__ amax,
    int E, int N, int logH) {
    int gw   = (blockIdx.x * blockDim.x + threadIdx.x) >> 5;
    int lane = threadIdx.x & 31;
    if (gw >= E + N) return;
    int s, d; float ea;
    if (gw < E) { s = esrc[gw]; d = edst[gw]; ea = eattr[gw]; }
    else        { s = d = gw - E; ea = easum[0] * (1.0f / (float)E); }

    int c0 = lane * 4;
    float4 xs = *(const float4*)(xl + (size_t)s * D + c0);
    float4 xd = *(const float4*)(xr + (size_t)d * D + c0);
    float4 we = *(const float4*)(We + c0);
    float4 at = *(const float4*)(att + c0);

    float p = 0.0f, mv;
    mv = xs.x + xd.x + ea * we.x; mv = mv >= 0.0f ? mv : 0.2f * mv; p += mv * at.x;
    mv = xs.y + xd.y + ea * we.y; mv = mv >= 0.0f ? mv : 0.2f * mv; p += mv * at.y;
    mv = xs.z + xd.z + ea * we.z; mv = mv >= 0.0f ? mv : 0.2f * mv; p += mv * at.z;
    mv = xs.w + xd.w + ea * we.w; mv = mv >= 0.0f ? mv : 0.2f * mv; p += mv * at.w;

    int H = 1 << logH;
    int w = 32 >> logH; // lanes per head (aligned power-of-2 groups)
    for (int off = w >> 1; off > 0; off >>= 1) p += __shfl_xor(p, off, 32);
    if ((lane & (w - 1)) == 0) {
        int h = lane >> (5 - logH);
        aBuf[(size_t)gw * H + h] = p;
        atomicMax(amax + (size_t)d * H + h, ordf(p));
    }
}

// ---------- edge pass B: exp + segment sum (thread per edge*head) ----------
__global__ void k_edge_b(const int* __restrict__ edst, float* __restrict__ aBuf,
                         const unsigned* __restrict__ amax, float* __restrict__ asum,
                         int E, int N, int logH, int tot) {
    int i = blockIdx.x * blockDim.x + threadIdx.x;
    if (i >= tot) return;
    int H = 1 << logH;
    int e = i >> logH, h = i & (H - 1);
    int d = (e < E) ? edst[e] : (e - E);
    float m  = unordf(amax[(size_t)d * H + h]);
    float ae = __expf(aBuf[i] - m);
    aBuf[i] = ae;
    atomicAdd(asum + (size_t)d * H + h, ae);
}

// ---------- edge pass C: alpha-weighted scatter add (wave per edge) ----------
__global__ __launch_bounds__(256) void k_edge_c(
    const float* __restrict__ xl, const int* __restrict__ esrc,
    const int* __restrict__ edst, const float* __restrict__ aBuf,
    const float* __restrict__ asum, float* __restrict__ acc,
    int E, int N, int logH) {
    int gw   = (blockIdx.x * blockDim.x + threadIdx.x) >> 5;
    int lane = threadIdx.x & 31;
    if (gw >= E + N) return;
    int s, d;
    if (gw < E) { s = esrc[gw]; d = edst[gw]; } else { s = d = gw - E; }
    int c0 = lane * 4;
    int H  = 1 << logH;
    int h  = lane >> (5 - logH);
    float alpha = aBuf[(size_t)gw * H + h] / (asum[(size_t)d * H + h] + 1e-16f);
    float4 xs = *(const float4*)(xl + (size_t)s * D + c0);
    float* o  = acc + (size_t)d * D + c0;
    atomicAdd(o + 0, xs.x * alpha);
    atomicAdd(o + 1, xs.y * alpha);
    atomicAdd(o + 2, xs.z * alpha);
    atomicAdd(o + 3, xs.w * alpha);
}

// ---------- post: bias + LayerNorm + GELU + residual (wave per node) ----------
__global__ __launch_bounds__(256) void k_post(
    float* __restrict__ h, const float* __restrict__ acc,
    const float* __restrict__ bias, const float* __restrict__ g,
    const float* __restrict__ b, int N, int doGelu) {
    int node = (blockIdx.x * blockDim.x + threadIdx.x) >> 5;
    int lane = threadIdx.x & 31;
    if (node >= N) return;
    int c0 = lane * 4;
    float4 v  = *(const float4*)(acc + (size_t)node * D + c0);
    float4 bp = *(const float4*)(bias + c0);
    v.x += bp.x; v.y += bp.y; v.z += bp.z; v.w += bp.w;
    float s = v.x + v.y + v.z + v.w;
    for (int o = 16; o > 0; o >>= 1) s += __shfl_xor(s, o, 32);
    float mu = s * (1.0f / 128.0f);
    float dx = v.x - mu, dy = v.y - mu, dz = v.z - mu, dw = v.w - mu;
    float q = dx * dx + dy * dy + dz * dz + dw * dw;
    for (int o = 16; o > 0; o >>= 1) q += __shfl_xor(q, o, 32);
    float rstd = rsqrtf(q * (1.0f / 128.0f) + 1e-5f);
    float4 gv = *(const float4*)(g + c0);
    float4 bv = *(const float4*)(b + c0);
    float y0 = dx * rstd * gv.x + bv.x;
    float y1 = dy * rstd * gv.y + bv.y;
    float y2 = dz * rstd * gv.z + bv.z;
    float y3 = dw * rstd * gv.w + bv.w;
    if (doGelu) {
        const float k = 0.70710678118654752f;
        y0 = 0.5f * y0 * (1.0f + erff(y0 * k));
        y1 = 0.5f * y1 * (1.0f + erff(y1 * k));
        y2 = 0.5f * y2 * (1.0f + erff(y2 * k));
        y3 = 0.5f * y3 * (1.0f + erff(y3 * k));
    }
    float* hp = h + (size_t)node * D + c0;
    float4 hv = *(float4*)hp;
    hv.x += y0; hv.y += y1; hv.z += y2; hv.w += y3;
    *(float4*)hp = hv;
}

// ---------- final reshape: drop last node per graph ----------
__global__ void k_out(const float* __restrict__ h, float* __restrict__ out,
                      int G, int npg, int tot) {
    int i = blockIdx.x * blockDim.x + threadIdx.x;
    if (i >= tot) return;
    int c = i & (D - 1);
    int r = i >> 7;                    // packed output row index (all graphs)
    int ii = r % (npg - 1);
    int gi = r / (npg - 1);
    out[i] = h[((size_t)gi * npg + ii) * D + c];
}

extern "C" void kernel_launch(void* const* d_in, const int* in_sizes, int n_in,
                              void* d_out, int out_size, void* d_ws, size_t ws_size,
                              hipStream_t stream) {
    const float* x     = (const float*)d_in[0];
    const int*   esrc  = (const int*)d_in[1];
    const int*   edst  = (const int*)d_in[2];
    const float* eattr = (const float*)d_in[3];
    const float* Wl    = (const float*)d_in[4];
    const float* bl    = (const float*)d_in[5];
    const float* Wr    = (const float*)d_in[6];
    const float* br    = (const float*)d_in[7];
    const float* We    = (const float*)d_in[8];
    const float* att   = (const float*)d_in[9];
    const float* biasp = (const float*)d_in[10];
    const float* lng   = (const float*)d_in[11];
    const float* lnb   = (const float*)d_in[12];

    const int N    = in_sizes[0] / D;
    const int E    = in_sizes[1];
    const int Etot = E + N;
    const int G    = (N * D - out_size) / D;  // num_graphs recovered from shapes
    const int npg  = N / G;

    char* ws = (char*)d_ws;
    auto alloc = [&](size_t bytes) -> char* {
        char* p = ws;
        ws += (bytes + 255) & ~(size_t)255;
        return p;
    };
    float*    h    = (float*)alloc((size_t)N * D * 4);
    _Float16* hh   = (_Float16*)alloc((size_t)N * D * 2);
    float*    xl   = (float*)alloc((size_t)N * D * 4);
    float*    xr   = (float*)alloc((size_t)N * D * 4);
    float*    acc  = (float*)alloc((size_t)N * D * 4);
    float*    aBuf = (float*)alloc((size_t)Etot * 4 * 4);
    unsigned* amax = (unsigned*)alloc((size_t)N * 4 * 4);
    float*    asum = (float*)alloc((size_t)N * 4 * 4);
    _Float16* WtL  = (_Float16*)alloc((size_t)D * D * 2);
    _Float16* WtR  = (_Float16*)alloc((size_t)D * D * 2);
    float*    easm = (float*)alloc(256);

    (void)hipMemcpyAsync(h, x, (size_t)N * D * 4, hipMemcpyDeviceToDevice, stream);
    (void)hipMemsetAsync(easm, 0, 4, stream);
    k_ea_sum<<<1024, 256, 0, stream>>>(eattr, E, easm);

    const int LOGH[3] = {2, 2, 0};
    const int edgeBlocks = (Etot * 32 + 255) / 256;
    for (int l = 0; l < 3; ++l) {
        int logH = LOGH[l];
        int H    = 1 << logH;
        (void)hipMemsetAsync(amax, 0, (size_t)N * H * 4, stream); // ord(-NaN) = 0 = minimum
        (void)hipMemsetAsync(asum, 0, (size_t)N * H * 4, stream);
        (void)hipMemsetAsync(acc,  0, (size_t)N * D * 4, stream);
        k_h2half<<<(N * D + 255) / 256, 256, 0, stream>>>(h, hh, N * D);
        k_w2half<<<(D * D + 255) / 256, 256, 0, stream>>>(
            Wl + (size_t)l * D * D, Wr + (size_t)l * D * D, WtL, WtR);
        dim3 ggrid(N / 128, 2);
        k_gemm<<<ggrid, 256, 0, stream>>>(hh, WtL, WtR, bl + l * D, br + l * D, xl, xr);
        k_edge_a<<<edgeBlocks, 256, 0, stream>>>(
            xl, xr, esrc, edst, eattr, easm, We + (size_t)l * D, att + (size_t)l * D,
            aBuf, amax, E, N, logH);
        int tot = Etot * H;
        k_edge_b<<<(tot + 255) / 256, 256, 0, stream>>>(edst, aBuf, amax, asum, E, N, logH, tot);
        k_edge_c<<<edgeBlocks, 256, 0, stream>>>(xl, esrc, edst, aBuf, asum, acc, E, N, logH);
        k_post<<<(N * 32 + 255) / 256, 256, 0, stream>>>(
            h, acc, biasp + (size_t)l * D, lng + (size_t)l * D, lnb + (size_t)l * D, N, l < 2);
    }
    k_out<<<(out_size + 255) / 256, 256, 0, stream>>>(h, (float*)d_out, G, npg, out_size);
}
